// CNF_46660524704252
// MI455X (gfx1250) — matrix operand order repaired and checked by reading, not compiled, not run on hardware
//
#include <hip/hip_runtime.h>

// MI455X / gfx1250 fused CNF dynamics kernel.
// wave32 WMMA f32<=f16 16x16x32 for all matmuls; all weights staged once in
// LDS (transposed f16) so every B fragment is a contiguous 2x ds_load_b128;
// branch-free tanh via v_exp_f32/v_rcp_f32; JVP trace with precomputed
// (1-x^2) terms in registers and packed-f16 tangent fragment builds.

typedef __attribute__((ext_vector_type(16))) _Float16 v16h;
typedef __attribute__((ext_vector_type(8)))  _Float16 h8;
typedef __attribute__((ext_vector_type(8)))  float    v8f;

#define DN 8
#define HN 64

__device__ __forceinline__ v8f wmma32(v16h a, v16h b, v8f c) {
  return __builtin_amdgcn_wmma_f32_16x16x32_f16(false, a, false, b, (short)0, c,
                                                false, false);
}

// Branch-free tanh: 1 - 2/(e^{2x}+1); saturation via IEEE inf/zero arithmetic.
__device__ __forceinline__ float fast_tanh(float x) {
  float e = __builtin_amdgcn_exp2f(x * 2.885390081777927f);  // e^{2x}
  return 1.f - 2.f * __builtin_amdgcn_rcpf(e + 1.f);
}

// 16-bit A-matrix 16x32 layout: lane (m = l&15, hi = l>>4), VGPR r holds the
// f16 pair at K = (r>=4?16:0) + hi*8 + (r&3)*2  (+kt*32 for the K-tile).
__device__ __forceinline__ int a_kbase(int r, int hi) {
  return ((r & 4) << 2) + hi * 8 + ((r & 3) << 1);
}

__device__ __forceinline__ v16h joinA(h8 lo, h8 hh) {
  return __builtin_shufflevector(lo, hh, 0, 1, 2, 3, 4, 5, 6, 7, 8, 9, 10, 11,
                                 12, 13, 14, 15);
}

// A fragment = two contiguous 16B LDS loads (ds_load_b128).
__device__ __forceinline__ v16h ldsA(const _Float16* X, int kt, int m, int hi) {
  const _Float16* base = X + m * HN + kt * 32 + hi * 8;
  return joinA(*(const h8*)(base), *(const h8*)(base + 16));
}

// B fragment from a transposed f16 LDS weight (WT[n*ldk + k]): two 16B runs
// at kbase and kbase+16 (2x ds_load_b128). kbase in {0,32}+hi*8.
__device__ __forceinline__ v16h ldsB(const _Float16* WT, int ldk, int kbase,
                                     int nn) {
  const _Float16* base = WT + nn * ldk + kbase;
  return joinA(*(const h8*)(base), *(const h8*)(base + 16));
}

__global__ void __launch_bounds__(128)
cnf_fused(const float* __restrict__ t_p, const float* __restrict__ z,
          const float* __restrict__ w1_0, const float* __restrict__ b1_0,
          const float* __restrict__ w1_1, const float* __restrict__ b1_1,
          const float* __restrict__ w1_2, const float* __restrict__ b1_2,
          const float* __restrict__ w1_o, const float* __restrict__ b1_o,
          const float* __restrict__ w2_0, const float* __restrict__ b2_0,
          const float* __restrict__ w2_1, const float* __restrict__ b2_1,
          const float* __restrict__ w2_2, const float* __restrict__ b2_2,
          const float* __restrict__ w2_o, const float* __restrict__ b2_o,
          float* __restrict__ out, int Bn) {
  __shared__ __align__(32) _Float16 WT11[HN * HN];  // w1_1^T
  __shared__ __align__(32) _Float16 WT12[HN * HN];  // w1_2^T
  __shared__ __align__(32) _Float16 WT21[HN * HN];  // w2_1^T
  __shared__ __align__(32) _Float16 WT22[HN * HN];  // w2_2^T
  __shared__ __align__(32) _Float16 WT10[HN * 32];  // w1_0^T, K padded 9->32
  __shared__ __align__(32) _Float16 WT20[HN * 32];  // w2_0^T, K padded 9->32
  __shared__ __align__(32) _Float16 R1h[DN * HN];   // w1_0 rows 1..8 (f16)
  __shared__ __align__(32) _Float16 XBUF[4][4][16 * HN];  // wave: X1,X2,X3,TT
  __shared__ float WO[HN * DN];                     // w1_o (f32)

  const int tid = threadIdx.x;

  // ---- cooperative one-time weight staging (transposed, f16) ----
#pragma unroll
  for (int i = 0; i < 32; ++i) {
    int e = tid + i * 128;
    int k = e >> 6, n = e & 63;
    WT11[n * HN + k] = (_Float16)w1_1[e];
    WT12[n * HN + k] = (_Float16)w1_2[e];
    WT21[n * HN + k] = (_Float16)w2_1[e];
    WT22[n * HN + k] = (_Float16)w2_2[e];
  }
#pragma unroll
  for (int i = 0; i < 16; ++i) {
    int e = tid + i * 128;  // 0..2047
    int k = e & 31, n = e >> 5;
    float v1 = w1_0[(k < 9 ? k : 0) * HN + n];
    float v2 = w2_0[(k < 9 ? k : 0) * HN + n];
    WT10[n * 32 + k] = (k < 9) ? (_Float16)v1 : (_Float16)0.f;
    WT20[n * 32 + k] = (k < 9) ? (_Float16)v2 : (_Float16)0.f;
  }
#pragma unroll
  for (int i = 0; i < 4; ++i) {
    int e = tid + i * 128;
    R1h[e] = (_Float16)w1_0[HN + e];  // rows 1..8 of w1_0
    WO[e] = w1_o[e];
  }
  __syncthreads();

  const int lane = tid & 31;
  const int wave = tid >> 5;
  const int m = lane & 15;   // A-layout row; also C/B-layout column (n)
  const int hi = lane >> 4;  // A-layout K-half; B-layout K-half; C M-half
  const int kbB = hi * 8;    // per-lane B-fragment K offset
  const int row0 = blockIdx.x * 64 + wave * 16;

  _Float16* X1 = XBUF[wave][0];
  _Float16* X2 = XBUF[wave][1];
  _Float16* X3 = XBUF[wave][2];
  _Float16* TT = XBUF[wave][3];

  const float t = t_p[0];

  // ---- shared layer-0 input A fragment: s = [t, z] padded K=9 -> 32 ----
  const float* zr = z + (size_t)(row0 + m) * DN;
  v16h a0;
#pragma unroll
  for (int r = 0; r < 8; ++r) {
    int k0 = a_kbase(r, hi);
#pragma unroll
    for (int p = 0; p < 2; ++p) {
      int k = k0 + p;
      float zv = zr[(k - 1) & 7];  // in-bounds for any k; discarded if unused
      float v = (k == 0) ? t : ((k <= DN) ? zv : 0.f);
      a0[2 * r + p] = (_Float16)v;
    }
  }

  // ================= net1 forward =================
#pragma unroll
  for (int nt = 0; nt < 4; ++nt) {
    int nn = nt * 16 + m;
    v8f c = {};
    c = wmma32(a0, ldsB(WT10, 32, kbB, nn), c);
    float bv = b1_0[nn];
#pragma unroll
    for (int r = 0; r < 8; ++r)
      X1[(hi * 8 + r) * HN + nn] = (_Float16)fast_tanh(c[r] + bv);
  }
  {
    v16h xa0 = ldsA(X1, 0, m, hi), xa1 = ldsA(X1, 1, m, hi);
#pragma unroll
    for (int nt = 0; nt < 4; ++nt) {
      int nn = nt * 16 + m;
      v8f c = {};
      c = wmma32(xa0, ldsB(WT11, HN, 0 + kbB, nn), c);
      c = wmma32(xa1, ldsB(WT11, HN, 32 + kbB, nn), c);
      float bv = b1_1[nn];
#pragma unroll
      for (int r = 0; r < 8; ++r)
        X2[(hi * 8 + r) * HN + nn] = (_Float16)fast_tanh(c[r] + bv);
    }
  }
  {
    v16h xa0 = ldsA(X2, 0, m, hi), xa1 = ldsA(X2, 1, m, hi);
#pragma unroll
    for (int nt = 0; nt < 4; ++nt) {
      int nn = nt * 16 + m;
      v8f c = {};
      c = wmma32(xa0, ldsB(WT12, HN, 0 + kbB, nn), c);
      c = wmma32(xa1, ldsB(WT12, HN, 32 + kbB, nn), c);
      float bv = b1_2[nn];
#pragma unroll
      for (int r = 0; r < 8; ++r)
        X3[(hi * 8 + r) * HN + nn] = (_Float16)fast_tanh(c[r] + bv);
    }
  }
  // output layer: (16x64) @ (64x8 padded to 16) -> dz_dt
  {
    v16h xa0 = ldsA(X3, 0, m, hi), xa1 = ldsA(X3, 1, m, hi);
    // B fragment for w1_o (64x8) padded: build from LDS WO (f32)
    v16h b0, b1;
#pragma unroll
    for (int r = 0; r < 8; ++r) {
#pragma unroll
      for (int p = 0; p < 2; ++p) {
        int kA = a_kbase(r, hi) + p, kBk = 32 + a_kbase(r, hi) + p;
        b0[2 * r + p] = (m < DN) ? (_Float16)WO[kA * DN + m] : (_Float16)0.f;
        b1[2 * r + p] = (m < DN) ? (_Float16)WO[kBk * DN + m] : (_Float16)0.f;
      }
    }
    v8f c = {};
    c = wmma32(xa0, b0, c);
    c = wmma32(xa1, b1, c);
    if (m < DN) {
      float bv = b1_o[m];
#pragma unroll
      for (int r = 0; r < 8; ++r)
        out[(size_t)(row0 + hi * 8 + r) * DN + m] = c[r] + bv;
    }
  }

  // ================= net2 forward (g) =================
#pragma unroll
  for (int nt = 0; nt < 4; ++nt) {
    int nn = nt * 16 + m;
    v8f c = {};
    c = wmma32(a0, ldsB(WT20, 32, kbB, nn), c);
    float bv = b2_0[nn];
#pragma unroll
    for (int r = 0; r < 8; ++r)
      TT[(hi * 8 + r) * HN + nn] = (_Float16)fast_tanh(c[r] + bv);
  }
  {
    v16h xa0 = ldsA(TT, 0, m, hi), xa1 = ldsA(TT, 1, m, hi);  // reads precede
#pragma unroll
    for (int nt = 0; nt < 4; ++nt) {                          // in-place stores
      int nn = nt * 16 + m;
      v8f c = {};
      c = wmma32(xa0, ldsB(WT21, HN, 0 + kbB, nn), c);
      c = wmma32(xa1, ldsB(WT21, HN, 32 + kbB, nn), c);
      float bv = b2_1[nn];
#pragma unroll
      for (int r = 0; r < 8; ++r)
        TT[(hi * 8 + r) * HN + nn] = (_Float16)fast_tanh(c[r] + bv);
    }
  }
  float gacc[8];
#pragma unroll
  for (int r = 0; r < 8; ++r) gacc[r] = 0.f;
  {
    v16h xa0 = ldsA(TT, 0, m, hi), xa1 = ldsA(TT, 1, m, hi);
#pragma unroll
    for (int nt = 0; nt < 4; ++nt) {
      int nn = nt * 16 + m;
      v8f c = {};
      c = wmma32(xa0, ldsB(WT22, HN, 0 + kbB, nn), c);
      c = wmma32(xa1, ldsB(WT22, HN, 32 + kbB, nn), c);
      float bv = b2_2[nn];
      float wv = w2_o[nn];  // final 64->1 contraction done elementwise
#pragma unroll
      for (int r = 0; r < 8; ++r) gacc[r] += fast_tanh(c[r] + bv) * wv;
    }
  }
#pragma unroll
  for (int r = 0; r < 8; ++r) {
    gacc[r] += __shfl_xor(gacc[r], 1);
    gacc[r] += __shfl_xor(gacc[r], 2);
    gacc[r] += __shfl_xor(gacc[r], 4);
    gacc[r] += __shfl_xor(gacc[r], 8);
    gacc[r] += b2_o[0];
  }

  // ================= exact trace via 8 forward-mode JVPs =================
  // (1 - x1^2) as packed f16, in A-fragment half order (4x h8 = 16 VGPRs).
  h8 a1q[4];
#pragma unroll
  for (int s = 0; s < 4; ++s) {
    const _Float16* base = X1 + m * HN + s * 16 + hi * 8;
    h8 x = *(const h8*)base;
    a1q[s] = (_Float16)1.f - x * x;  // v_pk_fma_f16 path
  }
  // (1 - x2^2), (1 - x3^2) once in f32 registers, C-layout order.
  float s2[32], s3[32];
#pragma unroll
  for (int nt = 0; nt < 4; ++nt)
#pragma unroll
    for (int r = 0; r < 8; ++r) {
      int nn = nt * 16 + m;
      float x2v = (float)X2[(hi * 8 + r) * HN + nn];
      float x3v = (float)X3[(hi * 8 + r) * HN + nn];
      s2[nt * 8 + r] = 1.f - x2v * x2v;
      s3[nt * 8 + r] = 1.f - x3v * x3v;
    }

  float tr[8];
#pragma unroll
  for (int r = 0; r < 8; ++r) tr[r] = 0.f;
#pragma unroll 1
  for (int d = 0; d < DN; ++d) {
    // x1' = (1 - x1^2) * w1_0[1+d,:] built in A layout: 4x ds_load_b128 +
    // packed f16 muls.
    const _Float16* r1 = R1h + d * HN + hi * 8;
    h8 r0 = *(const h8*)(r1);
    h8 r16 = *(const h8*)(r1 + 16);
    h8 r32 = *(const h8*)(r1 + 32);
    h8 r48 = *(const h8*)(r1 + 48);
    v16h p0 = joinA(a1q[0] * r0, a1q[1] * r16);
    v16h p1 = joinA(a1q[2] * r32, a1q[3] * r48);
    // u2' = x1' @ w1_1 ; x2' = (1 - x2^2) * u2' -> scratch
#pragma unroll
    for (int nt = 0; nt < 4; ++nt) {
      int nn = nt * 16 + m;
      v8f c = {};
      c = wmma32(p0, ldsB(WT11, HN, 0 + kbB, nn), c);
      c = wmma32(p1, ldsB(WT11, HN, 32 + kbB, nn), c);
#pragma unroll
      for (int r = 0; r < 8; ++r)
        TT[(hi * 8 + r) * HN + nn] = (_Float16)(s2[nt * 8 + r] * c[r]);
    }
    // u3' = x2' @ w1_2 ; tr += (1 - x3^2) * u3' . w1_o[:, d]
    v16h q0 = ldsA(TT, 0, m, hi), q1 = ldsA(TT, 1, m, hi);
#pragma unroll
    for (int nt = 0; nt < 4; ++nt) {
      int nn = nt * 16 + m;
      v8f c = {};
      c = wmma32(q0, ldsB(WT12, HN, 0 + kbB, nn), c);
      c = wmma32(q1, ldsB(WT12, HN, 32 + kbB, nn), c);
      float wv = WO[nn * DN + d];
#pragma unroll
      for (int r = 0; r < 8; ++r) tr[r] += s3[nt * 8 + r] * (c[r] * wv);
    }
  }
#pragma unroll
  for (int r = 0; r < 8; ++r) {
    tr[r] += __shfl_xor(tr[r], 1);
    tr[r] += __shfl_xor(tr[r], 2);
    tr[r] += __shfl_xor(tr[r], 4);
    tr[r] += __shfl_xor(tr[r], 8);
  }

  // dlogp = g - trace ; write dlogp and g (one writer lane per half-wave)
  if (m == 0) {
#pragma unroll
    for (int r = 0; r < 8; ++r) {
      int row = row0 + hi * 8 + r;
      out[(size_t)Bn * 8 + row] = gacc[r] - tr[r];
      out[(size_t)Bn * 9 + row] = gacc[r];
    }
  }
}

extern "C" void kernel_launch(void* const* d_in, const int* in_sizes, int n_in,
                              void* d_out, int out_size, void* d_ws,
                              size_t ws_size, hipStream_t stream) {
  const float* t    = (const float*)d_in[0];
  const float* z    = (const float*)d_in[1];
  // d_in[2] = logp_z (zeros, unused by the dynamics)
  const float* w1_0 = (const float*)d_in[3];
  const float* b1_0 = (const float*)d_in[4];
  const float* w1_1 = (const float*)d_in[5];
  const float* b1_1 = (const float*)d_in[6];
  const float* w1_2 = (const float*)d_in[7];
  const float* b1_2 = (const float*)d_in[8];
  const float* w1_o = (const float*)d_in[9];
  const float* b1_o = (const float*)d_in[10];
  const float* w2_0 = (const float*)d_in[11];
  const float* b2_0 = (const float*)d_in[12];
  const float* w2_1 = (const float*)d_in[13];
  const float* b2_1 = (const float*)d_in[14];
  const float* w2_2 = (const float*)d_in[15];
  const float* b2_2 = (const float*)d_in[16];
  const float* w2_o = (const float*)d_in[17];
  const float* b2_o = (const float*)d_in[18];
  float* out = (float*)d_out;

  int Bn = in_sizes[1] / DN;     // 262144
  int blocks = Bn / 64;          // 64 rows per 128-thread block (4 waves x 16)
  hipLaunchKernelGGL(cnf_fused, dim3(blocks), dim3(128), 0, stream,
                     t, z, w1_0, b1_0, w1_1, b1_1, w1_2, b1_2, w1_o, b1_o,
                     w2_0, b2_0, w2_1, b2_1, w2_2, b2_2, w2_o, b2_o, out, Bn);
}
